// MultiHeadSelfAttention_90254442758704
// MI455X (gfx1250) — matrix operand order repaired and checked by reading, not compiled, not run on hardware
//
#include <hip/hip_runtime.h>

// ---------------------------------------------------------------------------
// MI455X (gfx1250) multi-head self-attention, bf16 WMMA pipeline.
//   B=2, S=2048, E=1024, H=16, D=64
// Round 3: async global->LDS staging with the correct builtin signature
// (generic v4i* pointers, per the round-2 diagnostic) + prefetch.
// ---------------------------------------------------------------------------

#define EMBED 1024
#define NHEAD 16
#define HDIM  64
#define BATCH 2
#define SEQ   2048
#define NTOK  (BATCH * SEQ)   // 4096

typedef __attribute__((ext_vector_type(16))) __bf16 v16bf;
typedef __attribute__((ext_vector_type(8)))  float  v8f;
typedef int v4i_b128 __attribute__((vector_size(16)));

union Frag16 {      // one WMMA 16-bit A/B operand: 16 bf16 = 8 VGPRs
    uint4 u[2];
    v16bf v;
};
union B16x8 {       // 8 bf16 <-> uint4
    uint4  u;
    __bf16 h[8];
};

// ---- gfx1250 async global->LDS copy (ASYNCcnt), with safe fallback --------
#if defined(__gfx1250__) && \
    __has_builtin(__builtin_amdgcn_global_load_async_to_lds_b128) && \
    __has_builtin(__builtin_amdgcn_s_wait_asynccnt)
#define HAVE_ASYNC_LDS 1
#else
#define HAVE_ASYNC_LDS 0
#endif

__device__ inline void async_copy16(const void* gsrc, void* ldst) {
#if HAVE_ASYNC_LDS
    __builtin_amdgcn_global_load_async_to_lds_b128(
        (v4i_b128*)gsrc, (v4i_b128*)ldst, /*offset=*/0, /*cpol=*/0);
#else
    *(uint4*)ldst = *(const uint4*)gsrc;
#endif
}

__device__ inline void async_wait0() {
#if HAVE_ASYNC_LDS
    __builtin_amdgcn_s_wait_asynccnt(0);
#endif
}

__device__ inline v8f v8f_zero() {
    v8f z;
#pragma unroll
    for (int i = 0; i < 8; ++i) z[i] = 0.0f;
    return z;
}

__device__ inline v8f wmma_bf16(v16bf a, v16bf b, v8f c) {
    // D = A(16x32 bf16) * B(32x16 bf16) + C(16x16 f32)
    return __builtin_amdgcn_wmma_f32_16x16x32_bf16(
        /*neg_a=*/false, a, /*neg_b=*/false, b,
        /*c_mod=*/(short)0, c, /*reuse_a=*/false, /*reuse_b=*/false);
}

// ---------------------------------------------------------------------------
// Kernel 1: fp32 -> bf16 elementwise (X)
// ---------------------------------------------------------------------------
__global__ __launch_bounds__(256) void cvt_f32_bf16(
    const float* __restrict__ in, __bf16* __restrict__ out) {
    size_t i = ((size_t)blockIdx.x * 256 + threadIdx.x) * 8;
    float4 a = *(const float4*)(in + i);
    float4 b = *(const float4*)(in + i + 4);
    B16x8 r;
    r.h[0] = (__bf16)a.x; r.h[1] = (__bf16)a.y;
    r.h[2] = (__bf16)a.z; r.h[3] = (__bf16)a.w;
    r.h[4] = (__bf16)b.x; r.h[5] = (__bf16)b.y;
    r.h[6] = (__bf16)b.z; r.h[7] = (__bf16)b.w;
    *(uint4*)(out + i) = r.u;
}

// ---------------------------------------------------------------------------
// Kernel 2: W[in][out] fp32 -> Wt[out][in] bf16 (so WMMA B-columns are
// contiguous per lane)
// ---------------------------------------------------------------------------
__global__ __launch_bounds__(1024) void transpose_to_bf16(
    const float* __restrict__ W, __bf16* __restrict__ Wt) {
    __shared__ float tile[32][33];
    const int tx = threadIdx.x, ty = threadIdx.y;
    const int o = blockIdx.x * 32 + tx;
    const int i = blockIdx.y * 32 + ty;
    tile[ty][tx] = W[(size_t)i * EMBED + o];
    __syncthreads();
    const int oo = blockIdx.x * 32 + ty;
    const int ii = blockIdx.y * 32 + tx;
    Wt[(size_t)oo * EMBED + ii] = (__bf16)tile[tx][ty];
}

// ---------------------------------------------------------------------------
// Kernel 3: GEMM  C[4096,1024] = A[4096,1024] @ Wt^T + bias
//   A bf16 row-major, Wt bf16 [N][K] (N-major).
//   mode 0: write bf16 scattered into [B,H,S,D]   (Q/K/V projections)
//   mode 1: write f32 row-major [NTOK][EMBED]     (output projection)
// Block tile 128x128, 8 waves (4x2), wave tile 32x64, K-step 32.
// Tiles staged with async global->LDS copies.
// ---------------------------------------------------------------------------
__global__ __launch_bounds__(256) void gemm_bf16(
    const __bf16* __restrict__ A, const __bf16* __restrict__ Bt,
    const float* __restrict__ bias, float* __restrict__ outF,
    __bf16* __restrict__ outQ, int mode) {
    __shared__ __align__(16) __bf16 As[128 * 32];
    __shared__ __align__(16) __bf16 Bs[128 * 32];

    const int tid  = threadIdx.x;
    const int lane = tid & 31;
    const int w    = tid >> 5;
    const int wm   = w >> 1;       // 0..3
    const int wn   = w & 1;        // 0..1
    const int m0   = blockIdx.y * 128;
    const int n0   = blockIdx.x * 128;

    v8f acc[2][4];
#pragma unroll
    for (int mt = 0; mt < 2; ++mt)
#pragma unroll
        for (int nt = 0; nt < 4; ++nt) acc[mt][nt] = v8f_zero();

    const int lrow   = tid >> 1;         // 0..127
    const int lchunk = (tid & 1) * 16;   // 0 or 16 (halves)
    const int c0     = (lane >> 4) * 8;  // A-frag K sub-chunk base
    const int kb16   = (lane >> 4) * 16; // unused; B-frags also use c0

    const __bf16* gA = A  + (size_t)(m0 + lrow) * EMBED + lchunk;
    const __bf16* gB = Bt + (size_t)(n0 + lrow) * EMBED + lchunk;
    __bf16* sA = As + lrow * 32 + lchunk;
    __bf16* sB = Bs + lrow * 32 + lchunk;

    for (int kk = 0; kk < EMBED; kk += 32) {
        // stage A and B tiles directly into LDS (no VGPR bounce)
        async_copy16(gA + kk,     sA);
        async_copy16(gA + kk + 8, sA + 8);
        async_copy16(gB + kk,     sB);
        async_copy16(gB + kk + 8, sB + 8);
        if (kk + 32 < EMBED) {   // pull next tiles toward L2 while we compute
            __builtin_prefetch(gA + kk + 32, 0, 1);
            __builtin_prefetch(gB + kk + 32, 0, 1);
        }
        async_wait0();
        __syncthreads();

        Frag16 af[2], fB[4];
#pragma unroll
        for (int mt = 0; mt < 2; ++mt) {
            const int r = wm * 32 + mt * 16 + (lane & 15);
            af[mt].u[0] = *(const uint4*)(As + r * 32 + c0);
            af[mt].u[1] = *(const uint4*)(As + r * 32 + 16 + c0);
        }
#pragma unroll
        for (int nt = 0; nt < 4; ++nt) {
            const int cc = wn * 64 + nt * 16 + (lane & 15);
            fB[nt].u[0] = *(const uint4*)(Bs + cc * 32 + c0);
            fB[nt].u[1] = *(const uint4*)(Bs + cc * 32 + 16 + c0);
        }
#pragma unroll
        for (int mt = 0; mt < 2; ++mt)
#pragma unroll
            for (int nt = 0; nt < 4; ++nt)
                acc[mt][nt] = wmma_bf16(af[mt].v, fB[nt].v, acc[mt][nt]);
        __syncthreads();
    }

    // epilogue: C/D layout -> VGPR v holds row v (lanes 0-15) / v+8 (16-31)
#pragma unroll
    for (int mt = 0; mt < 2; ++mt) {
#pragma unroll
        for (int nt = 0; nt < 4; ++nt) {
            const int colg = n0 + wn * 64 + nt * 16 + (lane & 15);
            const float bv = bias[colg];
#pragma unroll
            for (int v = 0; v < 8; ++v) {
                const int rowg = m0 + wm * 32 + mt * 16 + v + ((lane >> 4) << 3);
                const float val = acc[mt][nt][v] + bv;
                if (mode == 0) {
                    const int hh = colg >> 6, dd = colg & 63;
                    const int bb = rowg >> 11, ss = rowg & 2047;
                    outQ[(((size_t)(bb * NHEAD + hh)) * SEQ + ss) * HDIM + dd] = (__bf16)val;
                } else {
                    outF[(size_t)rowg * EMBED + colg] = val;
                }
            }
        }
    }
}

// ---------------------------------------------------------------------------
// Kernel 4: flash attention.  Q,K,V bf16 [B,H,S,D]; ctx bf16 [NTOK, EMBED].
// Block = 128 queries of one (b,h); 8 waves x 16 queries; 32 keys / iter.
// K tile staged async; V tile transposed through VGPRs into LDS.
// ---------------------------------------------------------------------------
__global__ __launch_bounds__(256) void attn_kernel(
    const __bf16* __restrict__ Q, const __bf16* __restrict__ K,
    const __bf16* __restrict__ V, __bf16* __restrict__ ctx) {
    __shared__ __align__(16) __bf16 Ks[32 * 64];       // [key][d]
    __shared__ __align__(16) __bf16 Vt[64 * 32];       // [d][key]
    __shared__ __align__(16) __bf16 Pb[8 * 16 * 32];   // per-wave P tiles

    const int tid  = threadIdx.x;
    const int lane = tid & 31;
    const int w    = tid >> 5;
    const int b    = blockIdx.z;
    const int h    = blockIdx.y;
    const int qb   = blockIdx.x;
    const size_t base = (size_t)(b * NHEAD + h) * SEQ * HDIM;

    const int q0   = qb * 128 + w * 16;
    const int hoff = (lane >> 4) * 8;    // A-frag K sub-chunk base
    const int kb16 = (lane >> 4) * 16;   // unused; B-frags also use hoff

    // Q fragments for the two d-steps (d 0..31, 32..63), resident all kernel
    Frag16 qf[2];
    {
        const int qrow = q0 + (lane & 15);
#pragma unroll
        for (int sd = 0; sd < 2; ++sd) {
            qf[sd].u[0] = *(const uint4*)(Q + base + (size_t)qrow * HDIM + sd * 32 + hoff);
            qf[sd].u[1] = *(const uint4*)(Q + base + (size_t)qrow * HDIM + sd * 32 + 16 + hoff);
        }
    }

    float m[8], l[8];
    v8f oacc[4];
#pragma unroll
    for (int v = 0; v < 8; ++v) { m[v] = -3.0e38f; l[v] = 0.0f; }
#pragma unroll
    for (int nt = 0; nt < 4; ++nt) oacc[nt] = v8f_zero();

    for (int kt = 0; kt < SEQ; kt += 32) {
        {   // stage K row-major (async) and V transposed (through VGPRs)
            async_copy16(K + base + (size_t)kt * HDIM + tid * 8, Ks + tid * 8);
            B16x8 vv;
            vv.u = *(const uint4*)(V + base + (size_t)kt * HDIM + tid * 8);
            const int key = tid >> 3, db = (tid & 7) * 8;
#pragma unroll
            for (int j = 0; j < 8; ++j) Vt[(db + j) * 32 + key] = vv.h[j];
            if (kt + 32 < SEQ) {
                __builtin_prefetch(K + base + (size_t)(kt + 32) * HDIM + tid * 8, 0, 1);
                __builtin_prefetch(V + base + (size_t)(kt + 32) * HDIM + tid * 8, 0, 1);
            }
        }
        async_wait0();
        __syncthreads();

        // scores: 2 key sub-tiles x 2 d-steps = 4 WMMAs
        v8f sc[2];
        sc[0] = v8f_zero(); sc[1] = v8f_zero();
#pragma unroll
        for (int nt = 0; nt < 2; ++nt) {
            const int key = nt * 16 + (lane & 15);
#pragma unroll
            for (int sd = 0; sd < 2; ++sd) {
                Frag16 kf;
                kf.u[0] = *(const uint4*)(Ks + key * 64 + sd * 32 + hoff);
                kf.u[1] = *(const uint4*)(Ks + key * 64 + sd * 32 + 16 + hoff);
                sc[nt] = wmma_bf16(qf[sd].v, kf.v, sc[nt]);
            }
        }

        // online softmax (row reductions across each 16-lane group)
        float p0[8], p1[8], scl[8];
#pragma unroll
        for (int v = 0; v < 8; ++v) {
            const float s0 = sc[0][v] * 0.125f;   // 1/sqrt(64)
            const float s1 = sc[1][v] * 0.125f;
            float rm = fmaxf(s0, s1);
#pragma unroll
            for (int mask = 1; mask <= 8; mask <<= 1)
                rm = fmaxf(rm, __shfl_xor(rm, mask, 32));
            const float mn = fmaxf(m[v], rm);
            const float sv = __expf(m[v] - mn);
            const float e0 = __expf(s0 - mn);
            const float e1 = __expf(s1 - mn);
            float rs = e0 + e1;
#pragma unroll
            for (int mask = 1; mask <= 8; mask <<= 1)
                rs += __shfl_xor(rs, mask, 32);
            l[v] = l[v] * sv + rs;
            m[v] = mn;
            scl[v] = sv; p0[v] = e0; p1[v] = e1;
        }
#pragma unroll
        for (int nt = 0; nt < 4; ++nt)
#pragma unroll
            for (int v = 0; v < 8; ++v) oacc[nt][v] *= scl[v];

        // P: D-layout -> A-layout via per-wave LDS scratch
        const int pbase = w * 512;
#pragma unroll
        for (int v = 0; v < 8; ++v) {
            const int r = v + ((lane >> 4) << 3);
            Pb[pbase + r * 32 + (lane & 15)]      = (__bf16)p0[v];
            Pb[pbase + r * 32 + 16 + (lane & 15)] = (__bf16)p1[v];
        }
        asm volatile("s_wait_dscnt 0" ::: "memory");   // wave-local LDS RAW

        Frag16 pf;
        {
            const int pr = lane & 15;
            pf.u[0] = *(const uint4*)(Pb + pbase + pr * 32 + hoff);
            pf.u[1] = *(const uint4*)(Pb + pbase + pr * 32 + 16 + hoff);
        }

        // P @ V: 4 WMMAs (d columns 0..63)
#pragma unroll
        for (int nt = 0; nt < 4; ++nt) {
            const int dc = nt * 16 + (lane & 15);
            Frag16 vf;
            vf.u[0] = *(const uint4*)(Vt + dc * 32 + hoff);
            vf.u[1] = *(const uint4*)(Vt + dc * 32 + 16 + hoff);
            oacc[nt] = wmma_bf16(pf.v, vf.v, oacc[nt]);
        }
        __syncthreads();   // protect Ks/Vt before next stage
    }

    // normalize and write context [tok][E]
#pragma unroll
    for (int nt = 0; nt < 4; ++nt) {
#pragma unroll
        for (int v = 0; v < 8; ++v) {
            const int rq = q0 + v + ((lane >> 4) << 3);
            const size_t tok = (size_t)b * SEQ + rq;
            const int col = h * HDIM + nt * 16 + (lane & 15);
            ctx[tok * EMBED + col] = (__bf16)(oacc[nt][v] / l[v]);
        }
    }
}

// ---------------------------------------------------------------------------
// Host launcher
// ---------------------------------------------------------------------------
extern "C" void kernel_launch(void* const* d_in, const int* in_sizes, int n_in,
                              void* d_out, int out_size, void* d_ws, size_t ws_size,
                              hipStream_t stream) {
    (void)in_sizes; (void)n_in; (void)out_size; (void)ws_size;

    const float* x  = (const float*)d_in[0];
    const float* wq = (const float*)d_in[1];
    const float* bq = (const float*)d_in[2];
    const float* wk = (const float*)d_in[3];
    const float* bk = (const float*)d_in[4];
    const float* wv = (const float*)d_in[5];
    const float* bv = (const float*)d_in[6];
    const float* wo = (const float*)d_in[7];
    const float* bo = (const float*)d_in[8];
    float* out = (float*)d_out;

    char* ws = (char*)d_ws;
    const size_t MB = 1u << 20;
    __bf16* Xb  = (__bf16*)(ws);              // 8 MiB; reused as context later
    __bf16* WT  = (__bf16*)(ws + 8 * MB);     // 4 x 2 MiB (Wq,Wk,Wv,Wo transposed)
    __bf16* Qb  = (__bf16*)(ws + 16 * MB);    // 8 MiB
    __bf16* Kb  = (__bf16*)(ws + 24 * MB);    // 8 MiB
    __bf16* Vb  = (__bf16*)(ws + 32 * MB);    // 8 MiB
    const size_t WSZ = (size_t)EMBED * EMBED; // elements per weight

    // 1) X -> bf16
    cvt_f32_bf16<<<(NTOK * EMBED) / (256 * 8), 256, 0, stream>>>(x, Xb);

    // 2) transpose weights -> bf16 [out][in]
    dim3 tg(EMBED / 32, EMBED / 32), tb(32, 32);
    transpose_to_bf16<<<tg, tb, 0, stream>>>(wq, WT + 0 * WSZ);
    transpose_to_bf16<<<tg, tb, 0, stream>>>(wk, WT + 1 * WSZ);
    transpose_to_bf16<<<tg, tb, 0, stream>>>(wv, WT + 2 * WSZ);
    transpose_to_bf16<<<tg, tb, 0, stream>>>(wo, WT + 3 * WSZ);

    // 3) Q/K/V projections (scatter into [B,H,S,D] bf16)
    dim3 gg(EMBED / 128, NTOK / 128);
    gemm_bf16<<<gg, 256, 0, stream>>>(Xb, WT + 0 * WSZ, bq, nullptr, Qb, 0);
    gemm_bf16<<<gg, 256, 0, stream>>>(Xb, WT + 1 * WSZ, bk, nullptr, Kb, 0);
    gemm_bf16<<<gg, 256, 0, stream>>>(Xb, WT + 2 * WSZ, bv, nullptr, Vb, 0);

    // 4) flash attention -> context bf16 [NTOK][EMBED] (reuses Xb storage)
    __bf16* Ctx = Xb;
    attn_kernel<<<dim3(SEQ / 128, NHEAD, BATCH), 256, 0, stream>>>(Qb, Kb, Vb, Ctx);

    // 5) output projection -> f32 d_out
    gemm_bf16<<<gg, 256, 0, stream>>>(Ctx, WT + 3 * WSZ, bo, out, nullptr, 1);
}
